// Photoreceptor_Block_24601572671769
// MI455X (gfx1250) — compile-verified, matrix operand order checked by reading, not run on hardware
//
#include <hip/hip_runtime.h>
#include <hip/hip_bf16.h>

typedef __attribute__((ext_vector_type(16))) _Float16 v16h;
typedef __attribute__((ext_vector_type(8)))  _Float16 v8h;
typedef __attribute__((ext_vector_type(8)))  float    v8f;

#define NPIX 16384   // B*H*W = 4*64*64
#define CH   256
#define HP   66      // padded H/W

union V16U { v16h v; v8h h[2]; };

// ---------------------------------------------------------------------------
// Pack a K x N (K=256) f32 weight matrix into WMMA B-operand lane layout (f16).
// Element B[k][n] = src[k*rowStride + (n-n0)*colStride], only for n in [n0,n0+ncols).
// Packed layout: ((kt*NT + nt)*32 + lane)*16 + e  (16 contiguous halves per lane).
// lane<16:  n = nt*16+lane,    k = kt*32 + (e<8 ? e   : e+8)   (K 0..7,16..23)
// lane>=16: n = nt*16+lane-16, k = kt*32 + (e<8 ? e+8 : e+16)  (K 8..15,24..31)
// ---------------------------------------------------------------------------
__global__ void pack_b(const float* __restrict__ src, _Float16* __restrict__ dst,
                       int NT, int n0, int ncols, int rowStride, int colStride) {
  int idx = blockIdx.x * blockDim.x + threadIdx.x;
  int total = 8 * NT * 512;                 // KT=8 (K=256)
  if (idx >= total) return;
  int e    = idx & 15;
  int lane = (idx >> 4) & 31;
  int tile = idx >> 9;
  int nt = tile % NT;
  int kt = tile / NT;
  int n = nt * 16 + (lane & 15);
  int k = kt * 32 + ((lane < 16) ? ((e < 8) ? e : (e + 8))
                                 : ((e < 8) ? (e + 8) : (e + 16)));
  if (n < n0 || n >= n0 + ncols) return;    // padding columns stay zero (pre-memset)
  dst[idx] = (_Float16)src[(size_t)k * rowStride + (size_t)(n - n0) * colStride];
}

// ---------------------------------------------------------------------------
// Templated WMMA GEMM: C[16384 x (NT*16)] = sum_taps A_tap[16384 x 256] * B_tap + bias
// One wave computes a 16x64 tile (4 f32 accumulators), K-loop = 8 x (16x16x32).
// NT is always a multiple of 4 and the grid covers exactly NT/4 n-groups, so
// there are NO tile guards -> no divergent control flow around WMMA (EXEC all-1s).
//   APAD: A rows addressed through zero-padded (4,66,66,256) image at tap (dy,dx)
//   OPAD: result written into padded (4,66,66,256) interior (DCN gather image)
// Unrolling capped (taps x1, kt x2) to keep live VGPRs low -> no spills.
// ---------------------------------------------------------------------------
template <int NT, int TAPS, int TAPBASE, int APAD, int OPAD>
__global__ __launch_bounds__(256) void gemm_wmma(const _Float16* __restrict__ A,
                                                 const _Float16* __restrict__ Bp,
                                                 const float* __restrict__ bias,
                                                 float* __restrict__ Co) {
  int gw = (blockIdx.x * blockDim.x + threadIdx.x) >> 5;
  int mt = gw & 1023;                 // 1024 M-tiles (16384/16)
  int ng = gw >> 10;                  // n-tile group (4 tiles each), ng < NT/4
  int lane  = threadIdx.x & 31;
  int mloc  = lane & 15;
  int khalf = (lane < 16) ? 0 : 8;
  int m0 = mt << 4;
  int b = m0 >> 12, h = (m0 >> 6) & 63, w0 = m0 & 63;  // 16 pixels share (b,h)

  v8f acc[4] = {};
#pragma unroll 1
  for (int t = 0; t < TAPS; t++) {
    int tt = TAPBASE + t;
    int dy = tt / 3, dx = tt - dy * 3;
    const _Float16* arow = APAD
        ? (A + ((size_t)((b * HP + h + dy) * HP + (w0 + mloc + dx))) * CH + khalf)
        : (A + ((size_t)(m0 + mloc)) * CH + khalf);
    // per-lane B pointer for this tap; tile (kt,nt) is at + ((kt*NT+nt)<<9) halves
    const _Float16* bl = Bp + ((size_t)t << 12) * NT + ((size_t)(ng << 2) << 9)
                            + ((size_t)lane << 4);
#pragma unroll 2
    for (int kt = 0; kt < 8; kt++) {
      V16U ua;
      ua.h[0] = *(const v8h*)(arow + kt * 32);        // K 0..7   (or 8..15)
      ua.h[1] = *(const v8h*)(arow + kt * 32 + 16);   // K 16..23 (or 24..31)
#pragma unroll
      for (int j = 0; j < 4; j++) {
        v16h bm = *(const v16h*)(bl + (((size_t)(kt * NT + j)) << 9));
        acc[j] = __builtin_amdgcn_wmma_f32_16x16x32_f16(
            false, ua.v, false, bm, (short)0, acc[j], false, false);
      }
    }
  }
  int ncol = lane & 15;
  constexpr int N = NT << 4;
#pragma unroll
  for (int j = 0; j < 4; j++) {
    int n = ((ng << 2) + j) * 16 + ncol;
    float bv = bias ? bias[n] : 0.f;
#pragma unroll
    for (int r = 0; r < 8; r++) {
      int ml = r + ((lane < 16) ? 0 : 8);             // D layout: VGPR r -> M=r / r+8
      float v = acc[j][r] + bv;
      if (OPAD)
        Co[((size_t)((b * HP + h + 1) * HP + (w0 + ml + 1))) * CH + n] = v;
      else
        Co[(size_t)(m0 + ml) * N + n] = v;
    }
  }
}

// --------- per-(b,c) spatial mean of NCHW input (for SE block) --------------
__global__ void spatial_mean(const float* __restrict__ x, float* __restrict__ pm) {
  __shared__ float sh[256];
  int bc = blockIdx.x;                          // 1024 = B*C
  const float* p = x + (size_t)bc * 4096;
  float s = 0.f;
  for (int i = threadIdx.x; i < 4096; i += 256) s += p[i];
  sh[threadIdx.x] = s; __syncthreads();
  for (int st = 128; st > 0; st >>= 1) {
    if (threadIdx.x < st) sh[threadIdx.x] += sh[threadIdx.x + st];
    __syncthreads();
  }
  if (threadIdx.x == 0) pm[bc] = sh[0] * (1.f / 4096.f);
}

// --------- SE gain + reflective feedback: scale=1+sigmoid(...), rbias=relu(t) ---
__global__ void se_kernel(const float* __restrict__ pm,
                          const float* __restrict__ g1w, const float* __restrict__ g1b,
                          const float* __restrict__ g2w, const float* __restrict__ g2b,
                          const float* __restrict__ tw,  const float* __restrict__ tb,
                          const float* __restrict__ refl,
                          float* __restrict__ scale, float* __restrict__ rbias) {
  __shared__ float hid[64];
  for (int b = 0; b < 4; b++) {
    if (threadIdx.x < 64) {
      float s = g1b[threadIdx.x];
      for (int k = 0; k < 256; k++) s += g1w[threadIdx.x * 256 + k] * pm[b * 256 + k];
      hid[threadIdx.x] = fmaxf(s, 0.f);
    }
    __syncthreads();
    int c = threadIdx.x;
    float s = g2b[c];
    for (int j = 0; j < 64; j++) s += g2w[c * 64 + j] * hid[j];
    float gain = 1.f / (1.f + expf(-s));
    scale[b * 256 + c] = 1.f + gain;
    rbias[b * 256 + c] = fmaxf(tw[c] * refl[b] + tb[c], 0.f);
    __syncthreads();
  }
}

// --------- NCHW f32 -> channels-last f16 (x) and padded channels-last f16 (xr) ---
__global__ void prep_act(const float* __restrict__ x, const float* __restrict__ scale,
                         const float* __restrict__ rbias,
                         _Float16* __restrict__ xh, _Float16* __restrict__ xrp) {
  size_t idx = (size_t)blockIdx.x * blockDim.x + threadIdx.x;  // 4*256*4096 elems, NCHW
  int b = (int)(idx >> 20);
  int c = (int)((idx >> 12) & 255);
  int hw = (int)(idx & 4095);
  int h = hw >> 6, w = hw & 63;
  float v = x[idx];
  size_t m = ((size_t)b << 12) + hw;
  xh[m * CH + c] = (_Float16)v;
  float xr = v * scale[b * 256 + c] + rbias[b * 256 + c];
  xrp[(((size_t)b * HP + h + 1) * HP + (w + 1)) * CH + c] = (_Float16)xr;
}

// --------- BN stats: per-channel sum / sumsq over 16384 rows (channels-last) ---
__global__ void stats256(const float* __restrict__ x, float* __restrict__ sum,
                         float* __restrict__ sumsq) {
  int c = threadIdx.x;                          // 256
  size_t r0 = (size_t)blockIdx.x * 64;          // 256 blocks * 64 rows
  float s = 0.f, q = 0.f;
  for (int i = 0; i < 64; i++) {
    float v = x[(r0 + i) * CH + c];
    s += v; q += v * v;
  }
  atomicAdd(&sum[c], s);
  atomicAdd(&sumsq[c], q);
}

__global__ void finalize_stats(const float* __restrict__ sum, const float* __restrict__ sumsq,
                               const float* __restrict__ g, const float* __restrict__ b,
                               float* __restrict__ sc, float* __restrict__ sh) {
  int c = threadIdx.x;
  float mu = sum[c] * (1.f / 16384.f);
  float var = sumsq[c] * (1.f / 16384.f) - mu * mu;
  float s = g[c] * rsqrtf(var + 1e-5f);
  sc[c] = s;
  sh[c] = b[c] - mu * s;
}

// --------- BN apply + ReLU variants -----------------------------------------
__global__ void bn_relu_pad_h(const float* __restrict__ x, const float* __restrict__ sc,
                              const float* __restrict__ sh, _Float16* __restrict__ outp) {
  size_t idx = (size_t)blockIdx.x * blockDim.x + threadIdx.x;  // channels-last
  int c = (int)(idx & 255);
  size_t m = idx >> 8;
  int b = (int)(m >> 12), hw = (int)(m & 4095), h = hw >> 6, w = hw & 63;
  float y = fmaxf(x[idx] * sc[c] + sh[c], 0.f);
  outp[(((size_t)b * HP + h + 1) * HP + (w + 1)) * CH + c] = (_Float16)y;
}

__global__ void bn_relu_f32(const float* __restrict__ x, const float* __restrict__ sc,
                            const float* __restrict__ sh, float* __restrict__ out) {
  size_t idx = (size_t)blockIdx.x * blockDim.x + threadIdx.x;
  int c = (int)(idx & 255);
  out[idx] = fmaxf(x[idx] * sc[c] + sh[c], 0.f);
}

__global__ void bn_relu_h(const float* __restrict__ x, const float* __restrict__ sc,
                          const float* __restrict__ sh, _Float16* __restrict__ out) {
  size_t idx = (size_t)blockIdx.x * blockDim.x + threadIdx.x;
  int c = (int)(idx & 255);
  out[idx] = (_Float16)fmaxf(x[idx] * sc[c] + sh[c], 0.f);
}

// --------- fused depthwise 3x3 + LayerNorm + exact GELU (wave per pixel) -----
__global__ void dwln(const _Float16* __restrict__ xrp, const float* __restrict__ dww,
                     const float* __restrict__ dwb, const float* __restrict__ lng,
                     const float* __restrict__ lnb, _Float16* __restrict__ x1h) {
  int wv = (int)((blockIdx.x * blockDim.x + threadIdx.x) >> 5);  // pixel m
  int lane = threadIdx.x & 31;
  int b = wv >> 12, hw = wv & 4095, h = hw >> 6, w = hw & 63;
  const _Float16* base = xrp + (((size_t)b * HP + h) * HP + w) * CH;
  float v[8];
#pragma unroll
  for (int j = 0; j < 8; j++) {
    int c = lane + 32 * j;
    float acc = dwb[c];
#pragma unroll
    for (int ky = 0; ky < 3; ky++)
#pragma unroll
      for (int kx = 0; kx < 3; kx++)
        acc += dww[c * 9 + ky * 3 + kx] * (float)base[((size_t)ky * HP + kx) * CH + c];
    v[j] = acc;
  }
  float s = 0.f, q = 0.f;
#pragma unroll
  for (int j = 0; j < 8; j++) { s += v[j]; q += v[j] * v[j]; }
  for (int o = 16; o >= 1; o >>= 1) { s += __shfl_xor(s, o, 32); q += __shfl_xor(q, o, 32); }
  float mu = s * (1.f / 256.f);
  float var = q * (1.f / 256.f) - mu * mu;
  float rs = rsqrtf(var + 1e-5f);
#pragma unroll
  for (int j = 0; j < 8; j++) {
    int c = lane + 32 * j;
    float y = (v[j] - mu) * rs * lng[c] + lnb[c];
    y = 0.5f * y * (1.f + erff(y * 0.70710678118654752f));   // exact GELU
    x1h[(size_t)wv * CH + c] = (_Float16)y;
  }
}

// --------- offset bias + per-group softmax of mask logits --------------------
// praw has row stride 128 (N padded to 8 tiles); cols 0..71 offsets, 72..107 mask
__global__ void offmask(const float* __restrict__ raw, const float* __restrict__ offb,
                        const float* __restrict__ mskb, float* __restrict__ off,
                        float* __restrict__ mk) {
  int m = blockIdx.x * blockDim.x + threadIdx.x;
  if (m >= NPIX) return;
  const float* r = raw + (size_t)m * 128;
  for (int i = 0; i < 72; i++) off[(size_t)m * 72 + i] = r[i] + offb[i];
  for (int g = 0; g < 4; g++) {
    float l[9], mx = -1e30f;
    for (int p = 0; p < 9; p++) { l[p] = r[72 + g * 9 + p] + mskb[g * 9 + p]; mx = fmaxf(mx, l[p]); }
    float s = 0.f;
    for (int p = 0; p < 9; p++) { l[p] = expf(l[p] - mx); s += l[p]; }
    float inv = 1.f / s;
    for (int p = 0; p < 9; p++) mk[(size_t)m * 36 + g * 9 + p] = l[p] * inv;
  }
}

// --------- DCNv3 core: bilinear gather from padded xin ----------------------
// px = w + p/3 + off_x ; py = h + p%3 + off_y  (padded coords, derived from reference)
__global__ void dcn_kernel(const float* __restrict__ xinp, const float* __restrict__ off,
                           const float* __restrict__ mk, _Float16* __restrict__ outh) {
  int m = blockIdx.x;
  int t = threadIdx.x;                 // 256 = 4 groups x 64 channels
  int g = t >> 6, cc = t & 63;
  int b = m >> 12, hw = m & 4095, h = hw >> 6, w = hw & 63;
  const float* base = xinp + (size_t)b * HP * HP * CH + g * 64 + cc;
  const float* om = off + (size_t)m * 72 + g * 18;
  const float* mm = mk + (size_t)m * 36 + g * 9;
  float acc = 0.f;
#pragma unroll
  for (int p = 0; p < 9; p++) {
    float px = (float)(w + p / 3) + om[p * 2];
    float py = (float)(h + p % 3) + om[p * 2 + 1];
    float wt = mm[p];
    float x0f = floorf(px), y0f = floorf(py);
    float fx = px - x0f, fy = py - y0f;
    int x0 = (int)x0f, y0 = (int)y0f;
#pragma unroll
    for (int dy = 0; dy < 2; dy++) {
      int yi = y0 + dy;
      if (yi < 0 || yi >= HP) continue;
      float wy = dy ? fy : (1.f - fy);
#pragma unroll
      for (int dx = 0; dx < 2; dx++) {
        int xi = x0 + dx;
        if (xi < 0 || xi >= HP) continue;
        float wx = dx ? fx : (1.f - fx);
        acc += wt * wy * wx * base[((size_t)yi * HP + xi) * CH];
      }
    }
  }
  outh[(size_t)m * CH + t] = (_Float16)acc;
}

// --------- final: rbn2 + relu + darkness mix, write NCHW f32 -----------------
__global__ void final_mix(const float* __restrict__ rc, const float* __restrict__ sc,
                          const float* __restrict__ sh, const float* __restrict__ cone,
                          const float* __restrict__ dark, float* __restrict__ out) {
  size_t idx = (size_t)blockIdx.x * blockDim.x + threadIdx.x;  // NCHW
  int b = (int)(idx >> 20);
  int co = (int)((idx >> 12) & 255);
  int hw = (int)(idx & 4095);
  size_t m = ((size_t)b << 12) + hw;
  float rod = fmaxf(rc[m * CH + co] * sc[co] + sh[co], 0.f);
  float d = dark[b];
  out[idx] = (1.f - d) * rod + d * cone[m * CH + co];
}

// ===========================================================================
extern "C" void kernel_launch(void* const* d_in, const int* in_sizes, int n_in,
                              void* d_out, int out_size, void* d_ws, size_t ws_size,
                              hipStream_t stream) {
  const float* x     = (const float*)d_in[0];
  const float* dark  = (const float*)d_in[1];
  const float* refl  = (const float*)d_in[2];
  const float* c1_w  = (const float*)d_in[3];
  const float* c1_b  = (const float*)d_in[4];
  const float* cbn1g = (const float*)d_in[5];
  const float* cbn1b = (const float*)d_in[6];
  const float* c2_w  = (const float*)d_in[7];
  const float* c2_b  = (const float*)d_in[8];
  const float* cbn2g = (const float*)d_in[9];
  const float* cbn2b = (const float*)d_in[10];
  const float* g1_w  = (const float*)d_in[11];
  const float* g1_b  = (const float*)d_in[12];
  const float* g2_w  = (const float*)d_in[13];
  const float* g2_b  = (const float*)d_in[14];
  const float* t_w   = (const float*)d_in[15];
  const float* t_b   = (const float*)d_in[16];
  const float* dw_w  = (const float*)d_in[17];
  const float* dw_b  = (const float*)d_in[18];
  const float* ln_g  = (const float*)d_in[19];
  const float* ln_b  = (const float*)d_in[20];
  const float* off_w = (const float*)d_in[21];
  const float* off_b = (const float*)d_in[22];
  const float* msk_w = (const float*)d_in[23];
  const float* msk_b = (const float*)d_in[24];
  const float* in_w  = (const float*)d_in[25];
  const float* in_b  = (const float*)d_in[26];
  const float* out_w = (const float*)d_in[27];
  const float* out_b = (const float*)d_in[28];
  const float* rbn1g = (const float*)d_in[29];
  const float* rbn1b = (const float*)d_in[30];
  const float* rconw = (const float*)d_in[31];
  const float* rconb = (const float*)d_in[32];
  const float* rbn2g = (const float*)d_in[33];
  const float* rbn2b = (const float*)d_in[34];
  (void)in_sizes; (void)n_in; (void)out_size; (void)ws_size;

  char* ws = (char*)d_ws;
  // workspace layout (bytes, all 256-aligned)
  _Float16* Wp1   = (_Float16*)(ws + 0);          //  131072
  _Float16* Wp2   = (_Float16*)(ws + 131072);     // 1179648 (9 taps)
  _Float16* Wp3   = (_Float16*)(ws + 1310720);    //  131072
  _Float16* Wp4   = (_Float16*)(ws + 1441792);    //   65536 (N=128 padded)
  _Float16* Wp5   = (_Float16*)(ws + 1507328);    //  131072
  _Float16* Wp6   = (_Float16*)(ws + 1638400);    //  131072
  _Float16* x_h   = (_Float16*)(ws + 1769472);    // 8388608
  _Float16* xr_p  = (_Float16*)(ws + 10158080);   // 8921088 padded
  float*    c1out = (float*)   (ws + 19079168);   // 16777216 (reused: c2out, proj)
  _Float16* c1p   = (_Float16*)(ws + 35856384);   // 8921088 padded
  float*    cone  = (float*)   (ws + 44777472);   // 16777216
  float*    xinp  = (float*)   (ws + 61554688);   // 17842176 padded (reused: rc)
  _Float16* x1_h  = (_Float16*)(ws + 79396864);   // 8388608 (reused: rmid)
  float*    praw  = (float*)   (ws + 87785472);   // 8388608 (16384 x 128)
  float*    offs  = (float*)   (ws + 96174080);   // 4718592
  float*    mask  = (float*)   (ws + 100892672);  // 2359296
  float*    pmean = (float*)   (ws + 103251968);  // 4096
  float*    scale = (float*)   (ws + 103256064);  // 4096
  float*    rbias = (float*)   (ws + 103260160);  // 4096
  float*    stats = (float*)   (ws + 103264256);  // 4 BNs x 1024 floats
  float*    c2out = c1out;
  float*    proj  = c1out;
  float*    rc    = xinp;
  _Float16* dcn_h = x_h;
  _Float16* rmid  = x1_h;

  // zero padded buffers / padded weight / BN accumulators
  hipMemsetAsync(xr_p, 0, 8921088, stream);
  hipMemsetAsync(c1p,  0, 8921088, stream);
  hipMemsetAsync(xinp, 0, 17842176, stream);
  hipMemsetAsync(Wp4,  0, 65536, stream);
  hipMemsetAsync(stats, 0, 16384, stream);

  // ---- pack weights into WMMA B layout ----
  pack_b<<<256, 256, 0, stream>>>(c1_w, Wp1, 16, 0, 256, 1, 256);           // transpose (co,ci)
  for (int t = 0; t < 9; t++)                                                // 3x3 conv taps
    pack_b<<<256, 256, 0, stream>>>(c2_w + t, Wp2 + (size_t)t * 65536, 16, 0, 256, 9, 2304);
  pack_b<<<256, 256, 0, stream>>>(in_w,  Wp3, 16, 0, 256, 256, 1);          // (k,n) direct
  pack_b<<<128, 256, 0, stream>>>(off_w, Wp4, 8, 0, 72, 72, 1);             // cols 0..71
  pack_b<<<128, 256, 0, stream>>>(msk_w, Wp4, 8, 72, 36, 36, 1);            // cols 72..107
  pack_b<<<256, 256, 0, stream>>>(out_w, Wp5, 16, 0, 256, 256, 1);
  pack_b<<<256, 256, 0, stream>>>(rconw, Wp6, 16, 0, 256, 1, 256);          // transpose

  // ---- SE gain + reflective feedback ----
  spatial_mean<<<1024, 256, 0, stream>>>(x, pmean);
  se_kernel<<<1, 256, 0, stream>>>(pmean, g1_w, g1_b, g2_w, g2_b, t_w, t_b, refl, scale, rbias);
  prep_act<<<16384, 256, 0, stream>>>(x, scale, rbias, x_h, xr_p);

  // ---- Cone path ----
  gemm_wmma<16, 1, 4, 0, 0><<<512, 256, 0, stream>>>(x_h, Wp1, c1_b, c1out);   // 1x1 conv
  stats256<<<256, 256, 0, stream>>>(c1out, stats + 0, stats + 256);
  finalize_stats<<<1, 256, 0, stream>>>(stats + 0, stats + 256, cbn1g, cbn1b, stats + 512, stats + 768);
  bn_relu_pad_h<<<16384, 256, 0, stream>>>(c1out, stats + 512, stats + 768, c1p);
  gemm_wmma<16, 9, 0, 1, 0><<<512, 256, 0, stream>>>(c1p, Wp2, c2_b, c2out);   // 3x3 conv (9 taps)
  stats256<<<256, 256, 0, stream>>>(c2out, stats + 1024, stats + 1280);
  finalize_stats<<<1, 256, 0, stream>>>(stats + 1024, stats + 1280, cbn2g, cbn2b, stats + 1536, stats + 1792);
  bn_relu_f32<<<16384, 256, 0, stream>>>(c2out, stats + 1536, stats + 1792, cone);

  // ---- Rod path: DCNv3 ----
  gemm_wmma<16, 1, 4, 1, 1><<<512, 256, 0, stream>>>(xr_p, Wp3, in_b, xinp);   // in_proj -> padded
  dwln<<<2048, 256, 0, stream>>>(xr_p, dw_w, dw_b, ln_g, ln_b, x1_h);          // dw + LN + GELU
  gemm_wmma<8, 1, 4, 0, 0><<<256, 256, 0, stream>>>(x1_h, Wp4, nullptr, praw); // off|msk proj (N=128)
  offmask<<<64, 256, 0, stream>>>(praw, off_b, msk_b, offs, mask);
  dcn_kernel<<<16384, 256, 0, stream>>>(xinp, offs, mask, dcn_h);
  gemm_wmma<16, 1, 4, 0, 0><<<512, 256, 0, stream>>>(dcn_h, Wp5, out_b, proj); // out_proj
  stats256<<<256, 256, 0, stream>>>(proj, stats + 2048, stats + 2304);
  finalize_stats<<<1, 256, 0, stream>>>(stats + 2048, stats + 2304, rbn1g, rbn1b, stats + 2560, stats + 2816);
  bn_relu_h<<<16384, 256, 0, stream>>>(proj, stats + 2560, stats + 2816, rmid);
  gemm_wmma<16, 1, 4, 0, 0><<<512, 256, 0, stream>>>(rmid, Wp6, rconb, rc);    // rconv 1x1
  stats256<<<256, 256, 0, stream>>>(rc, stats + 3072, stats + 3328);
  finalize_stats<<<1, 256, 0, stream>>>(stats + 3072, stats + 3328, rbn2g, rbn2b, stats + 3584, stats + 3840);

  // ---- mix ----
  final_mix<<<16384, 256, 0, stream>>>(rc, stats + 3584, stats + 3840, cone, dark, (float*)d_out);
}
